// GNN_Arb_6227702579883
// MI455X (gfx1250) — matrix operand order repaired and checked by reading, not compiled, not run on hardware
//
#include <hip/hip_runtime.h>
#include <hip/hip_bf16.h>

// CDNA5 / gfx1250: wave32, WMMA 16x16x32 bf16 with f32 accumulate.
typedef __attribute__((ext_vector_type(16))) __bf16 v16bf;
typedef __attribute__((ext_vector_type(8)))  float  v8f;

#define DIMX 128      // feature dim
#define NPTS 128      // nodes per batch
#define HPITCH 136    // LDS row pitch in bf16 elems (272B, 16B aligned, conflict-breaking)

// ---------------------------------------------------------------------------
// init: x = embed[tok], W2 -> bf16 row-major (e rows, d contiguous)
// ---------------------------------------------------------------------------
__global__ __launch_bounds__(256) void k_init(const int* __restrict__ tok,
                                              const float* __restrict__ embed,
                                              const float* __restrict__ W2,
                                              float* __restrict__ x,
                                              __bf16* __restrict__ W2bf) {
    int idx = blockIdx.x * 256 + threadIdx.x;
    if (idx < 8 * NPTS * DIMX) {
        int bn = idx >> 7, d = idx & (DIMX - 1);
        x[idx] = embed[tok[bn] * DIMX + d];
    }
    if (idx < DIMX * DIMX) {
        W2bf[idx] = (__bf16)W2[idx];
    }
}

// ---------------------------------------------------------------------------
// hidden: hi[bn,e] = x[bn,:]·W1[e,0:128]; hj[bn,e] = x[bn,:]·W1[e,128:256]
// ---------------------------------------------------------------------------
__global__ __launch_bounds__(128) void k_hidden(const float* __restrict__ x,
                                                const float* __restrict__ W1,
                                                float* __restrict__ hi,
                                                float* __restrict__ hj) {
    int bn = blockIdx.x;          // b*128 + n
    int t  = threadIdx.x;         // output feature e
    __shared__ float xs[DIMX];
    xs[t] = x[bn * DIMX + t];
    __syncthreads();
    const float* w = W1 + t * (2 * DIMX);
    float a = 0.f, bb = 0.f;
    #pragma unroll 4
    for (int d = 0; d < DIMX; ++d) {
        float xv = xs[d];
        a  += xv * w[d];
        bb += xv * w[DIMX + d];
    }
    hi[bn * DIMX + t] = a;
    hj[bn * DIMX + t] = bb;
}

// ---------------------------------------------------------------------------
// main: per (b,i) workgroup.
//   H[j,d]  = bf16(relu(hi[b,i,d] + hj[b,j,d] + b1[d]))     (LDS, 128x128)
//   m[j,e]  = relu( (H @ W2^T)[j,e] + b2[e] )               (WMMA bf16->f32)
//   msg[e]  = sum_j A[b,j,i] * m[j,e];  x[b,i,e] += msg[e]  (deterministic)
// ---------------------------------------------------------------------------
__global__ __launch_bounds__(256) void k_main(const float* __restrict__ hi,
                                              const float* __restrict__ hj,
                                              const float* __restrict__ A,
                                              const float* __restrict__ b1,
                                              const float* __restrict__ b2,
                                              const __bf16* __restrict__ W2bf,
                                              float* __restrict__ x) {
    const int bi  = blockIdx.x;        // b*128 + i
    const int b   = bi >> 7;
    const int i   = bi & (NPTS - 1);
    const int tid = threadIdx.x;

    __shared__ __align__(16) __bf16 H[NPTS * HPITCH];
    __shared__ float hi_s[DIMX], b1_s[DIMX], b2_s[DIMX], Acol[NPTS];
    __shared__ float msg_part[8 * DIMX];

    if (tid < DIMX) {
        hi_s[tid] = hi[bi * DIMX + tid];
        b1_s[tid] = b1[tid];
        b2_s[tid] = b2[tid];
        Acol[tid] = A[(b * NPTS + tid) * NPTS + i];   // A[b, j, i]
    }
    __syncthreads();

    // Build H in bf16 (sum done in f32, single RNE rounding).
    const float* hjb = hj + b * (NPTS * DIMX);
    for (int idx = tid; idx < NPTS * DIMX; idx += 256) {
        int j = idx >> 7, d = idx & (DIMX - 1);
        float v = hi_s[d] + hjb[idx] + b1_s[d];
        v = v > 0.f ? v : 0.f;
        H[j * HPITCH + d] = (__bf16)v;
    }
    __syncthreads();

    const int wave = tid >> 5;         // M-block (16 rows of j) owned by wave
    const int lane = tid & 31;
    const int lhi  = lane >> 4;        // lane half
    const int llo  = lane & 15;
    const int jrow = wave * 16 + llo;  // A-operand row (M = lane%16)

    for (int nt = 0; nt < 8; ++nt) {
        v8f acc = {0.f, 0.f, 0.f, 0.f, 0.f, 0.f, 0.f, 0.f};
        const int e = nt * 16 + llo;   // B-operand column (N = lane%16)
        #pragma unroll
        for (int kb = 0; kb < 4; ++kb) {
            // A frag: K split {0-7,16-23} (lanes 0-15) / {8-15,24-31} (lanes 16-31)
            const int da = kb * 32 + 8 * lhi;
            union { v16bf v; uint4 u[2]; } af, bfm;
            af.u[0] = *reinterpret_cast<const uint4*>(&H[jrow * HPITCH + da]);
            af.u[1] = *reinterpret_cast<const uint4*>(&H[jrow * HPITCH + da + 16]);
            // B frag: B[k][n] = W2[e=n][d=k]; K split {0-7,16-23} (lanes 0-15) / {8-15,24-31} (lanes 16-31)
            const __bf16* wp = W2bf + e * DIMX + kb * 32 + 8 * lhi;
            bfm.u[0] = *reinterpret_cast<const uint4*>(wp);
            bfm.u[1] = *reinterpret_cast<const uint4*>(wp + 16);
            acc = __builtin_amdgcn_wmma_f32_16x16x32_bf16(
                false, af.v, false, bfm.v, (short)0, acc, false, false);
        }
        // Epilogue: D VGPR r -> (j = 16*wave + r + 8*lhi, e), fixed-order reduce
        float part = 0.f;
        #pragma unroll
        for (int r = 0; r < 8; ++r) {
            int j = wave * 16 + r + 8 * lhi;
            float mv = acc[r] + b2_s[e];
            mv = mv > 0.f ? mv : 0.f;
            part += Acol[j] * mv;
        }
        part += __shfl_xor(part, 16, 32);      // combine lane halves (same e)
        if (lhi == 0) msg_part[wave * DIMX + e] = part;
    }
    __syncthreads();
    if (tid < DIMX) {
        float s = 0.f;
        #pragma unroll
        for (int w = 0; w < 8; ++w) s += msg_part[w * DIMX + tid];
        x[bi * DIMX + tid] += s;
    }
}

// ---------------------------------------------------------------------------
// head: t = relu(z @ Wo1^T + bo1); y = t @ Wo2^T  (10 outputs per row)
// out layout: [out(8,10) | x_all(8,128,10)] flat; out row == x_all[b,0,:]
// ---------------------------------------------------------------------------
__global__ __launch_bounds__(128) void k_head(const float* __restrict__ x,
                                              const float* __restrict__ Wo1,
                                              const float* __restrict__ bo1,
                                              const float* __restrict__ Wo2,
                                              float* __restrict__ out) {
    int bn = blockIdx.x;
    int t  = threadIdx.x;
    __shared__ float z[DIMX], hl[DIMX];
    z[t] = x[bn * DIMX + t];
    __syncthreads();
    const float* w = Wo1 + t * DIMX;
    float a = bo1[t];
    #pragma unroll 4
    for (int d = 0; d < DIMX; ++d) a += z[d] * w[d];
    hl[t] = a > 0.f ? a : 0.f;
    __syncthreads();
    if (t < 10) {
        const float* w2 = Wo2 + t * DIMX;
        float y = 0.f;
        #pragma unroll 4
        for (int d = 0; d < DIMX; ++d) y += hl[d] * w2[d];
        int bb = bn >> 7, n = bn & (NPTS - 1);
        out[80 + bn * 10 + t] = y;                 // x_all
        if (n == 0) out[bb * 10 + t] = y;          // out
    }
}

// ---------------------------------------------------------------------------
extern "C" void kernel_launch(void* const* d_in, const int* in_sizes, int n_in,
                              void* d_out, int out_size, void* d_ws, size_t ws_size,
                              hipStream_t stream) {
    const int*   tok   = (const int*)  d_in[0];
    const float* A     = (const float*)d_in[1];
    const float* embed = (const float*)d_in[2];
    const float* W1    = (const float*)d_in[3];
    const float* b1    = (const float*)d_in[4];
    const float* W2    = (const float*)d_in[5];
    const float* b2    = (const float*)d_in[6];
    const float* Wo1   = (const float*)d_in[7];
    const float* bo1   = (const float*)d_in[8];
    const float* Wo2   = (const float*)d_in[9];
    float* out = (float*)d_out;

    // Workspace: x | hi | hj (f32, 512KB each) | W2bf (32KB)
    float*  x    = (float*)d_ws;
    float*  hi   = x  + 8 * NPTS * DIMX;
    float*  hj   = hi + 8 * NPTS * DIMX;
    __bf16* W2bf = (__bf16*)(hj + 8 * NPTS * DIMX);

    k_init<<<512, 256, 0, stream>>>(tok, embed, W2, x, W2bf);
    for (int r = 0; r < 5; ++r) {
        k_hidden<<<8 * NPTS, 128, 0, stream>>>(x, W1, hi, hj);
        k_main  <<<8 * NPTS, 256, 0, stream>>>(hi, hj, A, b1, b2, W2bf, x);
    }
    k_head<<<8 * NPTS, 128, 0, stream>>>(x, Wo1, bo1, Wo2, out);
}
